// autoencoder_gru_14929306320899
// MI455X (gfx1250) — matrix operand order repaired and checked by reading, not compiled, not run on hardware
//
#include <hip/hip_runtime.h>
#include <hip/hip_bf16.h>

// ---------------------------------------------------------------------------
// GRU autoencoder (S=4096, D=512, V=32000, DEC_H=50) for MI455X / gfx1250.
// Serial RNN -> persistent wave32 kernel with grid barriers; L2-BW bound ->
// bf16 weight streaming; fb-matvec (150x32000 per step) via
// v_wmma_f32_16x16x32_bf16 with A pre-swizzled to contiguous-k tiles (k = 16*(lane/16) + elem).
// ---------------------------------------------------------------------------

#define S      4096
#define DHID   512
#define VOC    32000
#define EMBD   10
#define DECH   50
#define DEC3   150
#define NBLK   32
#define NTHR   512
#define NWAVE  16
#define KCHUNKS 1000          // 32000 / 32
#define RTILES  10            // 150 rows padded to 160 -> 10 tiles of 16
#define STRIPS  52
#define CPS     20            // k-chunks per strip (50 strips cover 1000)

typedef __attribute__((ext_vector_type(16))) __bf16 v16bf;
typedef __attribute__((ext_vector_type(8)))  float  v8f;

// workspace byte offsets (total ~19.7 MB)
#define OFF_BAR      0u          // 2 x u32 grid-barrier state
#define OFF_INVZ     64u         // f32 1/Z of previous step
#define OFF_H0       256u        // 2 x 512 f32 (ping-pong)
#define OFF_H1       4352u       // 2 x 512 f32
#define OFF_HD1      8448u       // 64 f32 decoder h1
#define OFF_HD2      8704u       // 64 f32 decoder h2
#define OFF_GICONST  8960u       // 160 f32 (dense part of decoder gi + bias)
#define OFF_ZPART    10240u      // 512 f32 per-wave softmax partials
#define OFF_GIACCP   12288u      // 52*160 f32 per-strip WMMA partials
#define OFF_EF32     65536u      // 32000 f32  exp(logits)
#define OFF_EBF      196608u     // 32000 bf16 exp(logits)
#define OFF_WENC     262144u     // 3 * 1536*512 bf16 (Whh0, Wih1, Whh1)
#define OFF_W2BF     5242880u    // 32000*64 bf16 (rows padded 50->64)
#define OFF_DWFB     9437184u    // 10*1000*32*16 bf16 WMMA A tiles

__device__ __forceinline__ unsigned short f2bf(float f) {
  unsigned u = __float_as_uint(f);
  u += 0x7fffu + ((u >> 16) & 1u);          // round to nearest even
  return (unsigned short)(u >> 16);
}
__device__ __forceinline__ float bflo(unsigned u) { return __uint_as_float(u << 16); }
__device__ __forceinline__ float bfhi(unsigned u) { return __uint_as_float(u & 0xffff0000u); }
__device__ __forceinline__ float sigmoidf(float x) { return 1.f / (1.f + __expf(-x)); }

__device__ __forceinline__ float wred32(float v) {
#pragma unroll
  for (int o = 16; o > 0; o >>= 1) v += __shfl_xor(v, o, 32);
  return v;
}

// deterministic sense-reversing grid barrier (NBLK resident blocks)
__device__ __forceinline__ void gridbar(unsigned* bar) {
  __threadfence();
  __syncthreads();
  if (threadIdx.x == 0) {
    volatile unsigned* cnt = bar;
    volatile unsigned* gen = bar + 1;
    unsigned g = *gen;
    if (atomicAdd(bar, 1u) == NBLK - 1u) {
      *cnt = 0u;
      __threadfence();
      atomicAdd(bar + 1, 1u);
    } else {
      while (*gen == g) __builtin_amdgcn_s_sleep(1);
    }
  }
  __syncthreads();
  __threadfence();
}

// dot(row of 512 bf16 weights, 512 f32 in LDS), reduced across the wave
__device__ __forceinline__ float dot512(const unsigned* __restrict__ wrow,
                                        const float* __restrict__ h, int lane) {
  float s = 0.f;
#pragma unroll
  for (int i = 0; i < 8; ++i) {
    int d = lane + 32 * i;                  // coalesced dword (bf16 pair) loads
    unsigned u = wrow[d];
    s = fmaf(bflo(u), h[2 * d], s);
    s = fmaf(bfhi(u), h[2 * d + 1], s);
  }
  return wred32(s);
}

// ---------------------------------------------------------------------------
// Prep: bf16 conversions, WMMA A-tile swizzle of dWih1[:,512:], state zeroing.
// ---------------------------------------------------------------------------
__global__ void prep_kernel(const float* __restrict__ eWhh0, const float* __restrict__ eWih1,
                            const float* __restrict__ eWhh1, const float* __restrict__ dWih1,
                            const float* __restrict__ W2, char* __restrict__ ws) {
  const long gid = (long)blockIdx.x * blockDim.x + threadIdx.x;
  const long gsz = (long)gridDim.x * blockDim.x;

  // encoder recurrent weights -> bf16 row-major
  unsigned short* wenc = (unsigned short*)(ws + OFF_WENC);
  for (long i = gid; i < 3L * 786432L; i += gsz) {
    long m = i / 786432L, r = i % 786432L;
    const float* src = (m == 0) ? eWhh0 : (m == 1) ? eWih1 : eWhh1;
    wenc[i] = f2bf(src[r]);
  }
  // W2 -> bf16, rows padded 50 -> 64 for 128B-per-row streaming
  unsigned short* w2bf = (unsigned short*)(ws + OFF_W2BF);
  for (long i = gid; i < 32000L * 64L; i += gsz) {
    long v = i >> 6, k = i & 63;
    w2bf[i] = f2bf((k < DECH) ? W2[v * DECH + k] : 0.f);
  }
  // dWih1 fb-columns -> WMMA A tiles in contiguous k order (B uses the same):
  // lane m holds K in {0..15}; lane m+16 holds K in {16..31}
  unsigned short* dwfb = (unsigned short*)(ws + OFF_DWFB);
  for (long i = gid; i < (long)RTILES * KCHUNKS * 32 * 16; i += gsz) {
    int halfIdx = (int)(i & 15);
    int lane = (int)((i >> 4) & 31);
    long cc = (i >> 9) % KCHUNKS;
    long rt = (i >> 9) / KCHUNKS;
    int mi = lane & 15;
    int ko = ((lane >> 4) << 4) + halfIdx;
    long m = rt * 16 + mi;
    long k = cc * 32 + ko;
    float v = (m < DEC3) ? dWih1[m * 32512L + 512L + k] : 0.f;
    dwfb[i] = f2bf(v);
  }
  // zero recurrent state / partial-sum slots / barrier
  float* f;
  f = (float*)(ws + OFF_H0);     for (long i = gid; i < 1024; i += gsz) f[i] = 0.f;
  f = (float*)(ws + OFF_H1);     for (long i = gid; i < 1024; i += gsz) f[i] = 0.f;
  f = (float*)(ws + OFF_HD1);    for (long i = gid; i < 64;   i += gsz) f[i] = 0.f;
  f = (float*)(ws + OFF_HD2);    for (long i = gid; i < 64;   i += gsz) f[i] = 0.f;
  f = (float*)(ws + OFF_ZPART);  for (long i = gid; i < 512;  i += gsz) f[i] = 0.f;
  f = (float*)(ws + OFF_GIACCP); for (long i = gid; i < STRIPS * 160; i += gsz) f[i] = 0.f;
  if (gid == 0) {
    unsigned* bar = (unsigned*)(ws + OFF_BAR);
    bar[0] = 0u; bar[1] = 0u;
    *(float*)(ws + OFF_INVZ) = 0.f;
  }
}

// ---------------------------------------------------------------------------
// Persistent kernel: encoder scan, transition, decoder scan.
// ---------------------------------------------------------------------------
__global__ __launch_bounds__(NTHR, 1)
void rnn_persist(const int* __restrict__ ids, const float* __restrict__ emb,
                 const float* __restrict__ eWih0, const float* __restrict__ ebih0,
                 const float* __restrict__ ebhh0, const float* __restrict__ ebih1,
                 const float* __restrict__ ebhh1, const float* __restrict__ dWih1,
                 const float* __restrict__ dWhh1, const float* __restrict__ dbih1,
                 const float* __restrict__ dbhh1, const float* __restrict__ dWih2,
                 const float* __restrict__ dWhh2, const float* __restrict__ dbih2,
                 const float* __restrict__ dbhh2, const float* __restrict__ b2,
                 char* __restrict__ ws, float* __restrict__ out) {
  __shared__ float smem[1152];

  unsigned* bar = (unsigned*)(ws + OFF_BAR);
  float* h0buf = (float*)(ws + OFF_H0);
  float* h1buf = (float*)(ws + OFF_H1);
  float* hd1 = (float*)(ws + OFF_HD1);
  float* hd2 = (float*)(ws + OFF_HD2);
  float* giconst = (float*)(ws + OFF_GICONST);
  float* zpart = (float*)(ws + OFF_ZPART);
  float* giaccp = (float*)(ws + OFF_GIACCP);
  float* ef32 = (float*)(ws + OFF_EF32);
  unsigned short* ebf = (unsigned short*)(ws + OFF_EBF);
  const unsigned* whh0 = (const unsigned*)(ws + OFF_WENC);
  const unsigned* wih1 = (const unsigned*)(ws + OFF_WENC + 1572864u);
  const unsigned* whh1 = (const unsigned*)(ws + OFF_WENC + 2u * 1572864u);
  const unsigned short* w2bf = (const unsigned short*)(ws + OFF_W2BF);
  const unsigned short* atiles = (const unsigned short*)(ws + OFF_DWFB);
  float* invZp = (float*)(ws + OFF_INVZ);

  const int tid = threadIdx.x;
  const int lane = tid & 31;
  const int wv = tid >> 5;
  const int gwv = blockIdx.x * NWAVE + wv;       // 0..511 (one wave per h-element)
  const int gtid = blockIdx.x * NTHR + tid;      // 0..16383

  // ================= Encoder: 4096 serial steps, 2 grid barriers each ======
  for (int t = 0; t < S; ++t) {
    const int p = t & 1;
    const float* h0o = h0buf + p * DHID;
    float* h0n = h0buf + (p ^ 1) * DHID;
    const float* h1o = h1buf + p * DHID;
    float* h1n = h1buf + (p ^ 1) * DHID;

    // ---- layer 0: gh0 = Whh0 @ h0, gi0 from embedding (K=10, lane 0) ----
    if (tid < DHID) smem[tid] = h0o[tid];
    __syncthreads();
    {
      const int j = gwv;
      float ghr = dot512(whh0 + (size_t)(j)*256, smem, lane);
      float ghz = dot512(whh0 + (size_t)(j + DHID) * 256, smem, lane);
      float ghn = dot512(whh0 + (size_t)(j + 2 * DHID) * 256, smem, lane);
      if (lane == 0) {
        const float* x = emb + (size_t)ids[t] * EMBD;
        float gir = ebih0[j], giz = ebih0[j + DHID], gin = ebih0[j + 2 * DHID];
#pragma unroll
        for (int k = 0; k < EMBD; ++k) {
          float xv = x[k];
          gir = fmaf(eWih0[(size_t)(j)*EMBD + k], xv, gir);
          giz = fmaf(eWih0[(size_t)(j + DHID) * EMBD + k], xv, giz);
          gin = fmaf(eWih0[(size_t)(j + 2 * DHID) * EMBD + k], xv, gin);
        }
        ghr += ebhh0[j]; ghz += ebhh0[j + DHID]; ghn += ebhh0[j + 2 * DHID];
        float r = sigmoidf(gir + ghr);
        float z = sigmoidf(giz + ghz);
        float n = tanhf(gin + r * ghn);
        h0n[j] = (1.f - z) * n + z * smem[j];
      }
    }
    gridbar(bar);

    // ---- layer 1: gi1 = Wih1 @ h0_new, gh1 = Whh1 @ h1 ----
    if (tid < DHID) { smem[tid] = h0n[tid]; smem[DHID + tid] = h1o[tid]; }
    __syncthreads();
    {
      const int j = gwv;
      float gir = dot512(wih1 + (size_t)(j)*256, smem, lane);
      float giz = dot512(wih1 + (size_t)(j + DHID) * 256, smem, lane);
      float gin = dot512(wih1 + (size_t)(j + 2 * DHID) * 256, smem, lane);
      float ghr = dot512(whh1 + (size_t)(j)*256, smem + DHID, lane);
      float ghz = dot512(whh1 + (size_t)(j + DHID) * 256, smem + DHID, lane);
      float ghn = dot512(whh1 + (size_t)(j + 2 * DHID) * 256, smem + DHID, lane);
      if (lane == 0) {
        gir += ebih1[j]; giz += ebih1[j + DHID]; gin += ebih1[j + 2 * DHID];
        ghr += ebhh1[j]; ghz += ebhh1[j + DHID]; ghn += ebhh1[j + 2 * DHID];
        float r = sigmoidf(gir + ghr);
        float z = sigmoidf(giz + ghz);
        float n = tanhf(gin + r * ghn);
        h1n[j] = (1.f - z) * n + z * smem[DHID + j];
      }
    }
    gridbar(bar);
  }

  // ============ transition: gi_const = dWih1[:, :512] @ dense + bih1 ======
  const float* dense = h1buf + (S & 1) * DHID;
  if (blockIdx.x == 0) {
    if (tid < DHID) smem[tid] = dense[tid];
    __syncthreads();
    for (int m = tid; m < DEC3; m += NTHR) {
      float s = dbih1[m];
      const float* wr = dWih1 + (size_t)m * 32512;
      for (int k = 0; k < DHID; ++k) s = fmaf(wr[k], smem[k], s);
      giconst[m] = s;
    }
  }
  gridbar(bar);

  // ================= Decoder: 4096 serial steps, 3 grid barriers each =====
  const int rt = gwv % RTILES;
  const int strip = gwv / RTILES;
  const int c0 = strip * CPS;
  const int c1 = (c0 + CPS > KCHUNKS) ? KCHUNKS : (c0 + CPS);
  const bool bload = ((lane & 15) == 0);         // lanes 0/16 carry B column 0

  for (int t = 0; t < S; ++t) {
    // ---- D1: gi_fb partials = dWfb @ exp(logits_{t-1}) via WMMA bf16 ----
    if (t > 0) {
      v8f acc = {};
      v16bf b = {};          // zero once: non-loading lanes stay zero all loop
      for (int c = c0; c < c1; ++c) {
        const v16bf* ap =
            (const v16bf*)(atiles + ((size_t)(rt * KCHUNKS + c) * 32 + lane) * 16);
        __builtin_prefetch((const char*)ap + 1024, 0, 1);   // next 1KB A tile
        v16bf a = *ap;
        if (bload)
          b = *(const v16bf*)(ebf + c * 32 + ((lane >> 4) << 4));
        acc = __builtin_amdgcn_wmma_f32_16x16x32_bf16(false, a, false, b,
                                                      (short)0, acc, false, false);
      }
      if (bload) {                                          // col 0: lanes 0 / 16
        float* gp = giaccp + strip * 160 + rt * 16 + (lane >> 4) * 8;
#pragma unroll
        for (int r = 0; r < 8; ++r) gp[r] = acc[r];
      }
    }
    gridbar(bar);

    // ---- D3: block 0 — Z reduce, small GRU cells (150x50 matvecs) ----
    if (blockIdx.x == 0) {
      smem[tid] = zpart[tid];
      __syncthreads();
      for (int o = 256; o > 0; o >>= 1) {
        if (tid < o) smem[tid] += smem[tid + o];
        __syncthreads();
      }
      if (tid == 0) {
        float iz = (t == 0) ? 0.f : 1.f / smem[0];
        *invZp = iz;
        smem[1024] = iz;
      }
      __syncthreads();
      const float invZ = smem[1024];
      float* giL  = smem + 512;
      float* ghL  = smem + 672;
      float* h1L  = smem + 832;
      float* h1nL = smem + 896;
      float* h2L  = smem + 960;
      if (tid < 64) {
        h1L[tid] = (tid < DECH) ? hd1[tid] : 0.f;
        h2L[tid] = (tid < DECH) ? hd2[tid] : 0.f;
      }
      if (tid < DEC3) {
        float s = 0.f;
        for (int s2 = 0; s2 < STRIPS; ++s2) s += giaccp[s2 * 160 + tid];  // fixed order
        giL[tid] = giconst[tid] + invZ * s;
      }
      __syncthreads();
      if (tid < DEC3) {
        float s = dbhh1[tid];
        const float* wr = dWhh1 + tid * DECH;
#pragma unroll
        for (int k = 0; k < DECH; ++k) s = fmaf(wr[k], h1L[k], s);
        ghL[tid] = s;
      }
      __syncthreads();
      if (tid < DECH) {
        float r = sigmoidf(giL[tid] + ghL[tid]);
        float z = sigmoidf(giL[DECH + tid] + ghL[DECH + tid]);
        float n = tanhf(giL[2 * DECH + tid] + r * ghL[2 * DECH + tid]);
        float h = (1.f - z) * n + z * h1L[tid];
        h1nL[tid] = h; hd1[tid] = h;
      }
      __syncthreads();
      if (tid < DEC3) {
        float si = dbih2[tid], sh = dbhh2[tid];
        const float* wi = dWih2 + tid * DECH;
        const float* wh = dWhh2 + tid * DECH;
#pragma unroll
        for (int k = 0; k < DECH; ++k) {
          si = fmaf(wi[k], h1nL[k], si);
          sh = fmaf(wh[k], h2L[k], sh);
        }
        giL[tid] = si; ghL[tid] = sh;
      }
      __syncthreads();
      if (tid < DECH) {
        float r = sigmoidf(giL[tid] + ghL[tid]);
        float z = sigmoidf(giL[DECH + tid] + ghL[DECH + tid]);
        float n = tanhf(giL[2 * DECH + tid] + r * ghL[2 * DECH + tid]);
        hd2[tid] = (1.f - z) * n + z * h2L[tid];
      }
    }
    gridbar(bar);

    // ---- D5: out[t-1] = e*invZ, logits_t = W2@h2 + b2, e = exp(logits) ----
    {
      if (tid < 64) smem[tid] = (tid < DECH) ? hd2[tid] : 0.f;
      __syncthreads();
      const float invZ = *invZp;
      float zp = 0.f;
      for (int v = gtid; v < VOC; v += NBLK * NTHR) {
        float eo = ef32[v];
        if (t > 0) out[(size_t)(t - 1) * VOC + v] = eo * invZ;
        const unsigned* wr = (const unsigned*)(w2bf + (size_t)v * 64);
        float s = b2[v];
#pragma unroll
        for (int i = 0; i < 32; ++i) {
          unsigned u = wr[i];
          s = fmaf(bflo(u), smem[2 * i], s);
          s = fmaf(bfhi(u), smem[2 * i + 1], s);
        }
        float ev = __expf(s);
        ef32[v] = ev;
        ebf[v] = f2bf(ev);
        zp += ev;
      }
      zp = wred32(zp);
      if (lane == 0) zpart[gwv] = zp;
    }
    gridbar(bar);
  }

  // ---- final row: out[S-1] = e * 1/Z ----
  if (blockIdx.x == 0) {
    smem[tid] = zpart[tid];
    __syncthreads();
    for (int o = 256; o > 0; o >>= 1) {
      if (tid < o) smem[tid] += smem[tid + o];
      __syncthreads();
    }
    if (tid == 0) *invZp = 1.f / smem[0];
  }
  gridbar(bar);
  {
    const float invZ = *invZp;
    for (int v = gtid; v < VOC; v += NBLK * NTHR)
      out[(size_t)(S - 1) * VOC + v] = ef32[v] * invZ;
  }
}

// ---------------------------------------------------------------------------
extern "C" void kernel_launch(void* const* d_in, const int* in_sizes, int n_in,
                              void* d_out, int out_size, void* d_ws, size_t ws_size,
                              hipStream_t stream) {
  const int*   ids   = (const int*)  d_in[0];
  const float* emb   = (const float*)d_in[1];
  const float* eWih0 = (const float*)d_in[2];
  const float* eWhh0 = (const float*)d_in[3];
  const float* ebih0 = (const float*)d_in[4];
  const float* ebhh0 = (const float*)d_in[5];
  const float* eWih1 = (const float*)d_in[6];
  const float* eWhh1 = (const float*)d_in[7];
  const float* ebih1 = (const float*)d_in[8];
  const float* ebhh1 = (const float*)d_in[9];
  const float* dWih1 = (const float*)d_in[10];
  const float* dWhh1 = (const float*)d_in[11];
  const float* dbih1 = (const float*)d_in[12];
  const float* dbhh1 = (const float*)d_in[13];
  const float* dWih2 = (const float*)d_in[14];
  const float* dWhh2 = (const float*)d_in[15];
  const float* dbih2 = (const float*)d_in[16];
  const float* dbhh2 = (const float*)d_in[17];
  const float* W2    = (const float*)d_in[18];
  const float* b2    = (const float*)d_in[19];
  char* ws = (char*)d_ws;
  float* out = (float*)d_out;

  prep_kernel<<<1024, 256, 0, stream>>>(eWhh0, eWih1, eWhh1, dWih1, W2, ws);
  rnn_persist<<<NBLK, NTHR, 0, stream>>>(ids, emb, eWih0, ebih0, ebhh0, ebih1, ebhh1,
                                         dWih1, dWhh1, dbih1, dbhh1,
                                         dWih2, dWhh2, dbih2, dbhh2,
                                         b2, ws, out);
}
